// SparseMultiDense_69140383531521
// MI455X (gfx1250) — matrix operand order repaired and checked by reading, not compiled, not run on hardware
//
#include <hip/hip_runtime.h>

#define M_CNT 8
#define B_DIM 4096
#define D_IN  512
#define D_OUT 512

#define BM 128   // block rows (batch dim)
#define BN 128   // block cols (output dim)
#define KC 32    // K chunk per stage
#define LDP 40   // padded LDS row length in bf16 elements (80B stride)

typedef __attribute__((ext_vector_type(16))) __bf16 v16bf;
typedef __attribute__((ext_vector_type(8)))  __bf16 v8bf;
typedef __attribute__((ext_vector_type(4)))  __bf16 v4bf;
typedef __attribute__((ext_vector_type(8)))  float  v8f;
typedef __attribute__((ext_vector_type(4)))  float  v4f;

__global__ __launch_bounds__(256)
void batched_dense_relu_wmma(const float* __restrict__ inp,
                             const float* __restrict__ wgt,
                             const float* __restrict__ bias,
                             float* __restrict__ out) {
  // Double-buffered bf16 hi/lo planes: 4 arrays * 2 bufs * 128*40*2B = 80 KB LDS
  __shared__ __bf16 sAhi[2][BM * LDP];
  __shared__ __bf16 sAlo[2][BM * LDP];
  __shared__ __bf16 sBhi[2][BN * LDP];   // transposed: [n][k]
  __shared__ __bf16 sBlo[2][BN * LDP];

  const int m      = blockIdx.z;
  const int rowBlk = blockIdx.x * BM;
  const int colBlk = blockIdx.y * BN;
  const int tid    = threadIdx.x;
  const int lane   = tid & 31;
  const int wave   = tid >> 5;
  const int waveRow = (wave >> 1) * 32;  // 4 wave-rows of 32
  const int waveCol = (wave & 1) * 64;   // 2 wave-cols of 64

  const float* Abase = inp + (size_t)m * B_DIM * D_IN + (size_t)rowBlk * D_IN;
  const float* Wbase = wgt + (size_t)m * D_IN * D_OUT + colBlk;

  // Staging maps
  const int aRow = tid >> 3;          // 0..31 (+32*it), row of A chunk
  const int aCol = (tid & 7) * 4;     // 0,4,..,28 within KC
  const int wKq  = (tid & 7) * 4;     // k-quad base: 0..28 (low bits -> banks)
  const int wNq  = (tid >> 3) * 4;    // n-quad base: 0..124

  v8f acc[2][4];
#pragma unroll
  for (int i = 0; i < 2; ++i)
#pragma unroll
    for (int j = 0; j < 4; ++j)
      acc[i][j] = (v8f){0.f, 0.f, 0.f, 0.f, 0.f, 0.f, 0.f, 0.f};

  const int r16 = lane & 15;
  const int kh  = (lane >> 4) * 8;    // A fragment K-half select
  const int ks  = (lane >> 4) * 16;   // unused; B fragment reads also use kh

  // ---- pipeline helpers ----
  auto stageLoad = [&](int k0, v4f* aR, v4f* wR) {
#pragma unroll
    for (int it = 0; it < 4; ++it)   // A: 128x32, 4 rows' worth per thread
      aR[it] = *(const v4f*)(Abase + (size_t)(aRow + it * 32) * D_IN + (k0 + aCol));
#pragma unroll
    for (int e = 0; e < 4; ++e)      // W: 4x4 block at (k0+wKq, wNq)
      wR[e] = *(const v4f*)(Wbase + (size_t)(k0 + wKq + e) * D_OUT + wNq);
  };

  auto stageStore = [&](int p, const v4f* aR, const v4f* wR) {
#pragma unroll
    for (int it = 0; it < 4; ++it) {
      v4bf hi, lo;
#pragma unroll
      for (int e = 0; e < 4; ++e) {
        float x  = aR[it][e];
        __bf16 h = (__bf16)x;
        hi[e] = h;
        lo[e] = (__bf16)(x - (float)h);
      }
      *(v4bf*)&sAhi[p][(aRow + it * 32) * LDP + aCol] = hi;
      *(v4bf*)&sAlo[p][(aRow + it * 32) * LDP + aCol] = lo;
    }
    // register 4x4 transpose of W -> [n][k] bf16 planes, contiguous b64 stores
#pragma unroll
    for (int c = 0; c < 4; ++c) {
      v4bf hi, lo;
#pragma unroll
      for (int e = 0; e < 4; ++e) {
        float x  = wR[e][c];
        __bf16 h = (__bf16)x;
        hi[e] = h;
        lo[e] = (__bf16)(x - (float)h);
      }
      *(v4bf*)&sBhi[p][(wNq + c) * LDP + wKq] = hi;
      *(v4bf*)&sBlo[p][(wNq + c) * LDP + wKq] = lo;
    }
  };

  auto computeChunk = [&](int p) {
    v16bf aHi[2], aLo[2];
#pragma unroll
    for (int i = 0; i < 2; ++i) {
      const __bf16* pah = &sAhi[p][(waveRow + i * 16 + r16) * LDP];
      const __bf16* pal = &sAlo[p][(waveRow + i * 16 + r16) * LDP];
      v8bf h0 = *(const v8bf*)(pah + kh);
      v8bf h1 = *(const v8bf*)(pah + 16 + kh);
      v8bf l0 = *(const v8bf*)(pal + kh);
      v8bf l1 = *(const v8bf*)(pal + 16 + kh);
#pragma unroll
      for (int e = 0; e < 8; ++e) {
        aHi[i][e] = h0[e]; aHi[i][8 + e] = h1[e];
        aLo[i][e] = l0[e]; aLo[i][8 + e] = l1[e];
      }
    }
    v16bf bHi[4], bLo[4];
#pragma unroll
    for (int j = 0; j < 4; ++j) {
      const __bf16* pbh = &sBhi[p][(waveCol + j * 16 + r16) * LDP];
      const __bf16* pbl = &sBlo[p][(waveCol + j * 16 + r16) * LDP];
      v8bf h0 = *(const v8bf*)(pbh + kh);
      v8bf h1 = *(const v8bf*)(pbh + 16 + kh);
      v8bf l0 = *(const v8bf*)(pbl + kh);
      v8bf l1 = *(const v8bf*)(pbl + 16 + kh);
#pragma unroll
      for (int e = 0; e < 8; ++e) {
        bHi[j][e] = h0[e]; bHi[j][8 + e] = h1[e];
        bLo[j][e] = l0[e]; bLo[j][8 + e] = l1[e];
      }
    }
#pragma unroll
    for (int i = 0; i < 2; ++i) {
#pragma unroll
      for (int j = 0; j < 4; ++j) {
        acc[i][j] = __builtin_amdgcn_wmma_f32_16x16x32_bf16(
            false, aHi[i], false, bHi[j], (short)0, acc[i][j], false, false);
        acc[i][j] = __builtin_amdgcn_wmma_f32_16x16x32_bf16(
            false, aHi[i], false, bLo[j], (short)0, acc[i][j], false, false);
        acc[i][j] = __builtin_amdgcn_wmma_f32_16x16x32_bf16(
            false, aLo[i], false, bHi[j], (short)0, acc[i][j], false, false);
      }
    }
  };

  // ---- software pipeline: 1 barrier per K chunk, loads in flight over WMMA ----
  {
    v4f aR[4], wR[4];
    stageLoad(0, aR, wR);
    stageStore(0, aR, wR);
  }
  __syncthreads();
  int p = 0;
  for (int k0 = KC; k0 < D_IN; k0 += KC) {
    v4f aR[4], wR[4];
    stageLoad(k0, aR, wR);      // global loads issued before the WMMA block
    computeChunk(p);            // WMMA on current buffer
    stageStore(p ^ 1, aR, wR);  // convert + fill other buffer
    __syncthreads();
    p ^= 1;
  }
  computeChunk(p);              // drain last chunk

  // ---- Epilogue: bias + ReLU + store (C layout: VGPR r -> M=r / r+8 by lane half) ----
  const int rAdd = (lane >> 4) * 8;
#pragma unroll
  for (int j = 0; j < 4; ++j) {
    const int gn = colBlk + waveCol + j * 16 + r16;
    const float bv = bias[m * D_OUT + gn];
#pragma unroll
    for (int i = 0; i < 2; ++i) {
      const int gm0 = rowBlk + waveRow + i * 16 + rAdd;
      float* po = out + (size_t)m * B_DIM * D_OUT + (size_t)gm0 * D_OUT + gn;
#pragma unroll
      for (int r = 0; r < 8; ++r) {
        float v = acc[i][j][r] + bv;
        po[(size_t)r * D_OUT] = v > 0.f ? v : 0.f;
      }
    }
  }
}

extern "C" void kernel_launch(void* const* d_in, const int* in_sizes, int n_in,
                              void* d_out, int out_size, void* d_ws, size_t ws_size,
                              hipStream_t stream) {
  (void)in_sizes; (void)n_in; (void)out_size; (void)d_ws; (void)ws_size;
  const float* inp  = (const float*)d_in[0];
  const float* wgt  = (const float*)d_in[1];
  const float* bias = (const float*)d_in[2];
  float* out = (float*)d_out;
  dim3 grid(B_DIM / BM, D_OUT / BN, M_CNT);  // 32 x 4 x 8 = 1024 blocks
  batched_dense_relu_wmma<<<grid, dim3(256), 0, stream>>>(inp, wgt, bias, out);
}
